// FC_large_80092550135881
// MI455X (gfx1250) — 8-bit matrix operand order repaired and checked by reading, compiled, not run on hardware
//
#include <hip/hip_runtime.h>
#include <stdint.h>

typedef __attribute__((ext_vector_type(16))) int   v16i;
typedef __attribute__((ext_vector_type(8)))  float v8f;

#define BATCH 16384
#define IND   768
#define HID   4096
#define NOUT  10
#define XCOLS 784
#define BN_EPS 1e-5f

#define FP8_P1 ((uint8_t)0x38)
#define FP8_M1 ((uint8_t)0xB8)

#define TILE        128
#define LDS_STRIDE  144
#define TILE_BYTES  (TILE * LDS_STRIDE)
#define BUF_BYTES   (2 * TILE_BYTES)
#define SMEM_BYTES  (2 * BUF_BYTES)

__device__ __forceinline__ void async_ld16(uint32_t lds_off, const uint8_t* g) {
  asm volatile("global_load_async_to_lds_b128 %0, %1, off"
               :: "v"(lds_off), "v"(g) : "memory");
}

__device__ __forceinline__ v16i load_bfrag(const uint8_t* bp, int half) {
  v16i b;
#pragma unroll
  for (int q = 0; q < 4; ++q) {
    const int4 v = *(const int4*)(bp + (q << 5) + (half << 4));
    b[4 * q]     = v.x;
    b[4 * q + 1] = v.y;
    b[4 * q + 2] = v.z;
    b[4 * q + 3] = v.w;
  }
  return b;
}

__global__ void binarize_x_fp8(const float* __restrict__ x, uint8_t* __restrict__ out) {
  int idx = blockIdx.x * blockDim.x + threadIdx.x;
  if (idx >= BATCH * IND) return;
  int row = idx / IND;
  int col = idx - row * IND;
  out[idx] = (x[(size_t)row * XCOLS + col] >= 0.f) ? FP8_P1 : FP8_M1;
}

__global__ void binarize_fp8(const float* __restrict__ w, uint8_t* __restrict__ out, int n) {
  int idx = blockIdx.x * blockDim.x + threadIdx.x;
  if (idx >= n) return;
  out[idx] = (w[idx] >= 0.f) ? FP8_P1 : FP8_M1;
}

__global__ void __launch_bounds__(256)
bgemm_fp8_lds(const uint8_t* __restrict__ A, const uint8_t* __restrict__ W,
              const float* __restrict__ bias, const float* __restrict__ gamma,
              const float* __restrict__ beta, const float* __restrict__ mean,
              const float* __restrict__ var,
              uint8_t* __restrict__ out, int M, int N, int K) {
  extern __shared__ __align__(16) uint8_t smem[];

  const int t    = threadIdx.x;
  const int wv   = t >> 5;
  const int lane = t & 31;
  const int half = lane >> 4;
  const int lrow = lane & 15;

  const int mbase = blockIdx.x * TILE;
  const int nbase = blockIdx.y * TILE;

  const uint32_t lds0 = (uint32_t)(uintptr_t)smem;

  const int      srow = t >> 1;
  const int      sh   = (t & 1) << 6;
  const uint8_t* gA   = A + (size_t)(mbase + srow) * K + sh;
  const uint8_t* gW   = W + (size_t)(nbase + srow) * K + sh;
  const uint32_t lA   = lds0 + (uint32_t)(srow * LDS_STRIDE + sh);
  const uint32_t lB   = lA + TILE_BYTES;

  v8f acc[8];
#pragma unroll
  for (int j = 0; j < 8; ++j) acc[j] = (v8f){};

  const int nch = K >> 7;

#pragma unroll
  for (int i = 0; i < 4; ++i) async_ld16(lA + 16u * i, gA + 16 * i);
#pragma unroll
  for (int i = 0; i < 4; ++i) async_ld16(lB + 16u * i, gW + 16 * i);

  for (int c = 0; c < nch; ++c) {
    const int cur = c & 1;

    if (c + 1 < nch) {
      const int      k1 = (c + 1) << 7;
      const uint32_t st = (uint32_t)((cur ^ 1) * BUF_BYTES);
#pragma unroll
      for (int i = 0; i < 4; ++i) async_ld16(lA + st + 16u * i, gA + k1 + 16 * i);
#pragma unroll
      for (int i = 0; i < 4; ++i) async_ld16(lB + st + 16u * i, gW + k1 + 16 * i);
      asm volatile("s_wait_asynccnt 8" ::: "memory");
    } else {
      asm volatile("s_wait_asynccnt 0" ::: "memory");
    }
    __syncthreads();

    const uint8_t* As = smem + cur * BUF_BYTES;
    const uint8_t* Bs = As + TILE_BYTES;

    v16i a;
    {
      const uint8_t* ap = As + (wv * 16 + lrow) * LDS_STRIDE;
#pragma unroll
      for (int p = 0; p < 8; ++p) {
        const int2 v = *(const int2*)(ap + ((p >> 1) << 5) + (half << 4) + ((p & 1) << 3));
        a[2 * p]     = v.x;
        a[2 * p + 1] = v.y;
      }
    }

    const uint8_t* brow = Bs + lrow * LDS_STRIDE;
    v16i bc = load_bfrag(brow, half);
#pragma unroll
    for (int j = 0; j < 8; ++j) {
      v16i bn;
      if (j < 7) bn = load_bfrag(brow + (j + 1) * 16 * LDS_STRIDE, half);
      acc[j] = __builtin_amdgcn_wmma_f32_16x16x128_fp8_fp8(a, bc, (short)0, acc[j],
                                                           false, false);
      bc = bn;
    }

    __syncthreads();
  }

#pragma unroll
  for (int j = 0; j < 8; ++j) {
    const int   n   = nbase + (j << 4) + lrow;
    const float sc  = gamma[n] * rsqrtf(var[n] + BN_EPS);
    const float off = (bias[n] - mean[n]) * sc + beta[n];
#pragma unroll
    for (int r = 0; r < 8; ++r) {
      const int   m = mbase + (wv << 4) + (half << 3) + r;
      const float y = acc[j][r] * sc + off;
      out[(size_t)m * N + n] = (y >= 0.f) ? FP8_P1 : FP8_M1;
    }
  }
}

__global__ void __launch_bounds__(256)
layer4_logsoftmax(const uint8_t* __restrict__ A, const uint8_t* __restrict__ W,
                  const float* __restrict__ bias, const float* __restrict__ gamma,
                  const float* __restrict__ beta, const float* __restrict__ mean,
                  const float* __restrict__ var, float* __restrict__ out) {
  const int wid  = (blockIdx.x * blockDim.x + threadIdx.x) >> 5;
  const int lane = threadIdx.x & 31;
  if (wid >= BATCH) return;

  const uint8_t* arow = A + (size_t)wid * HID;

  int acc[NOUT];
#pragma unroll
  for (int o = 0; o < NOUT; ++o) acc[o] = 0;

  for (int it = 0; it < HID / 128; ++it) {
    const int kb = (it * 32 + lane) * 4;
    const int av = *(const int*)(arow + kb);
#pragma unroll
    for (int o = 0; o < NOUT; ++o) {
      const int wv = *(const int*)(W + (size_t)o * HID + kb);
      acc[o] += 4 - 2 * __popc((av ^ wv) & 0x80808080);
    }
  }

#pragma unroll
  for (int o = 0; o < NOUT; ++o) {
#pragma unroll
    for (int s = 16; s > 0; s >>= 1)
      acc[o] += __shfl_xor(acc[o], s, 32);
  }

  if (lane == 0) {
    float logits[NOUT];
    float mx = -3.0e38f;
#pragma unroll
    for (int o = 0; o < NOUT; ++o) {
      const float sc = gamma[o] * rsqrtf(var[o] + BN_EPS);
      const float y  = ((float)acc[o] + bias[o] - mean[o]) * sc + beta[o];
      logits[o] = y;
      mx = fmaxf(mx, y);
    }
    float s = 0.f;
#pragma unroll
    for (int o = 0; o < NOUT; ++o) s += expf(logits[o] - mx);
    const float lse = mx + logf(s);
    float* orow = out + (size_t)wid * NOUT;
#pragma unroll
    for (int o = 0; o < NOUT; ++o) orow[o] = logits[o] - lse;
  }
}

extern "C" void kernel_launch(void* const* d_in, const int* in_sizes, int n_in,
                              void* d_out, int out_size, void* d_ws, size_t ws_size,
                              hipStream_t stream) {
  (void)in_sizes; (void)n_in; (void)out_size; (void)ws_size;

  const float* x  = (const float*)d_in[0];
  const float* w[4];  const float* bi[4]; const float* g[4];
  const float* be[4]; const float* mn[4]; const float* vr[4];
  for (int i = 0; i < 4; ++i) {
    w[i]  = (const float*)d_in[1 + 6 * i + 0];
    bi[i] = (const float*)d_in[1 + 6 * i + 1];
    g[i]  = (const float*)d_in[1 + 6 * i + 2];
    be[i] = (const float*)d_in[1 + 6 * i + 3];
    mn[i] = (const float*)d_in[1 + 6 * i + 4];
    vr[i] = (const float*)d_in[1 + 6 * i + 5];
  }

  uint8_t* ws = (uint8_t*)d_ws;
  size_t off = 0;
  uint8_t* A0   = ws + off; off += (size_t)BATCH * IND;
  uint8_t* ACT1 = ws + off; off += (size_t)BATCH * HID;
  uint8_t* ACT2 = ws + off; off += (size_t)BATCH * HID;
  uint8_t* W1B  = ws + off; off += (size_t)HID * IND;
  uint8_t* W2B  = ws + off; off += (size_t)HID * HID;
  uint8_t* W3B  = ws + off; off += (size_t)HID * HID;
  uint8_t* W4B  = ws + off; off += (size_t)NOUT * HID;

  const int THR = 256;

  binarize_x_fp8<<<(BATCH * IND + THR - 1) / THR, THR, 0, stream>>>(x, A0);
  binarize_fp8<<<(HID * IND + THR - 1) / THR, THR, 0, stream>>>(w[0], W1B, HID * IND);
  binarize_fp8<<<(HID * HID + THR - 1) / THR, THR, 0, stream>>>(w[1], W2B, HID * HID);
  binarize_fp8<<<(HID * HID + THR - 1) / THR, THR, 0, stream>>>(w[2], W3B, HID * HID);
  binarize_fp8<<<(NOUT * HID + THR - 1) / THR, THR, 0, stream>>>(w[3], W4B, NOUT * HID);

  dim3 grid(BATCH / TILE, HID / TILE);

  bgemm_fp8_lds<<<grid, THR, SMEM_BYTES, stream>>>(A0,   W1B, bi[0], g[0], be[0], mn[0], vr[0],
                                                   ACT1, BATCH, HID, IND);
  bgemm_fp8_lds<<<grid, THR, SMEM_BYTES, stream>>>(ACT1, W2B, bi[1], g[1], be[1], mn[1], vr[1],
                                                   ACT2, BATCH, HID, HID);
  bgemm_fp8_lds<<<grid, THR, SMEM_BYTES, stream>>>(ACT2, W3B, bi[2], g[2], be[2], mn[2], vr[2],
                                                   ACT1, BATCH, HID, HID);

  layer4_logsoftmax<<<BATCH * 32 / THR, THR, 0, stream>>>(ACT1, W4B, bi[3], g[3], be[3],
                                                          mn[3], vr[3], (float*)d_out);
}
